// STMambaModel_33483565039905
// MI455X (gfx1250) — compile-verified, matrix operand order checked by reading, not run on hardware
//
#include <hip/hip_runtime.h>

typedef __attribute__((ext_vector_type(16))) __bf16 v16bf;
typedef __attribute__((ext_vector_type(8)))  float  v8f;

#define TOK 100000   // B*T*P tokens
#define HD  128      // hidden dim H
#define DIN 256      // d_inner = 2H
#define G3  384      // 3H (GRU gates)
#define PP  5000     // patches
#define TT  20       // time steps

// ---------------------------------------------------------------------------
// WMMA fragment loader. Tile is stored row-major (row, 32*k... arbitrary ld).
// Per CDNA5 ISA 7.12.2 (16-bit A 16x32):
//   lane L: row = L%16 ; lanes 0-15 hold K {0..7, 16..23}, lanes 16-31 hold
//   K {8..15, 24..31}.  B tiles stored as W row-major (N,K) load identically
//   (role of M becomes N), giving C[m,n] = sum_k A[m,k]*W[n,k].
// ---------------------------------------------------------------------------
__device__ __forceinline__ v16bf load_frag_rm(const __bf16* base, int ld, int lane) {
  const int row = lane & 15;
  const int kh  = (lane >> 4) << 3;           // 0 or 8
  const __bf16* p0 = base + row * ld + kh;
  v16bf f;
#pragma unroll
  for (int j = 0; j < 8; ++j) { f[j] = p0[j]; f[8 + j] = p0[16 + j]; }
  return f;
}

__device__ __forceinline__ float gelu_exact(float x) {
  return 0.5f * x * (1.0f + erff(x * 0.70710678118654752f));
}

// ---------------------------------------------------------------------------
// Generic GEMM: C[M,N] = act(A[M,K] @ W^T + bias), W bf16 row-major (N,K).
// Block tile 64x64, 256 threads = 8 waves (4 M-subtiles x 2 N-subtiles),
// each wave owns a 16x32 output strip (two 16x16 f32 accumulators).
// A is fp32 in global memory, converted to bf16 while staging into LDS.
// ---------------------------------------------------------------------------
template<int ACT>
__global__ __launch_bounds__(256)
void gemm_wmma_kernel(const float* __restrict__ A, long long ldA,
                      const __bf16* __restrict__ W, int K,
                      const float* __restrict__ bias,
                      float* __restrict__ C, long long ldC,
                      int M)
{
  __shared__ __bf16 At[64][32];
  __shared__ __bf16 Wt[64][32];
  const int tid = threadIdx.x, lane = tid & 31, wave = tid >> 5;
  const int mBase = blockIdx.x * 64;
  const int nBase = blockIdx.y * 64;
  const int wm = wave & 3, wn = wave >> 2;
  v8f acc0 = {0.f,0.f,0.f,0.f,0.f,0.f,0.f,0.f};
  v8f acc1 = {0.f,0.f,0.f,0.f,0.f,0.f,0.f,0.f};
  const int lrow = tid >> 2, lcol = (tid & 3) << 3;   // 64 rows x 4 segs of 8

  for (int kc = 0; kc < K; kc += 32) {
    __syncthreads();
    {
      const int gm = mBase + lrow;
      const float* src = A + (long long)gm * ldA + kc + lcol;
#pragma unroll
      for (int j = 0; j < 8; ++j)
        At[lrow][lcol + j] = (gm < M) ? (__bf16)src[j] : (__bf16)0.0f;
      const __bf16* wsrc = W + (long long)(nBase + lrow) * K + kc + lcol;
#pragma unroll
      for (int j = 0; j < 8; ++j) Wt[lrow][lcol + j] = wsrc[j];
    }
    __syncthreads();
    v16bf a  = load_frag_rm(&At[wm * 16][0], 32, lane);
    v16bf b0 = load_frag_rm(&Wt[wn * 32][0], 32, lane);
    v16bf b1 = load_frag_rm(&Wt[wn * 32 + 16][0], 32, lane);
    acc0 = __builtin_amdgcn_wmma_f32_16x16x32_bf16(false, a, false, b0,
                                                   (short)0, acc0, false, false);
    acc1 = __builtin_amdgcn_wmma_f32_16x16x32_bf16(false, a, false, b1,
                                                   (short)0, acc1, false, false);
  }

  // C layout: lane L -> col = L%16 ; element r -> row = r + 8*(L/16)
  const int n0 = nBase + wn * 32 + (lane & 15);
  const int n1 = n0 + 16;
  const int mB = mBase + wm * 16 + ((lane >> 4) << 3);
  const float bi0 = bias[n0], bi1 = bias[n1];
#pragma unroll
  for (int r = 0; r < 8; ++r) {
    const int m = mB + r;
    if (m < M) {
      float v0 = acc0[r] + bi0, v1 = acc1[r] + bi1;
      if (ACT) { v0 = gelu_exact(v0); v1 = gelu_exact(v1); }
      C[(long long)m * ldC + n0] = v0;
      C[(long long)m * ldC + n1] = v1;
    }
  }
}

// ---------------------------------------------------------------------------
// GRU recurrence with WMMA. One workgroup scans up to 32 sequences in
// lockstep: per step gh = h(32x128) @ Whh^T (384x128 bf16, cache-resident).
// 2x24 = 48 WMMA tiles / step, split over 8 waves (6 tiles each, K-loop of 4).
// h state lives in registers (16 values/thread) + bf16 LDS mirror for WMMA.
// token(seq, step) = seq*seqStride + step*stepStride ; reverse flips scan
// order (handles backward spatial direction incl. the un-reverse).
// ---------------------------------------------------------------------------
__global__ __launch_bounds__(256)
void gru_scan_wmma(const float* __restrict__ xi,          // [tokens][384]
                   const __bf16* __restrict__ Whh,        // [384][128] bf16
                   const float* __restrict__ bhh,         // [384]
                   float* __restrict__ out, int outLd, int outColOff,
                   int nSeq, int seqLen,
                   long long stepStride, long long seqStride,
                   int reverse)
{
  __shared__ __bf16 hbf[32][HD];
  __shared__ float  gh[32][G3];
  const int tid = threadIdx.x, lane = tid & 31, wave = tid >> 5;
  const int seq0 = blockIdx.x * 32;

  float hreg[16];
#pragma unroll
  for (int i = 0; i < 16; ++i) hreg[i] = 0.0f;
  for (int i = tid; i < 32 * HD; i += 256) (&hbf[0][0])[i] = (__bf16)0.0f;

  for (int s = 0; s < seqLen; ++s) {
    const int step = reverse ? (seqLen - 1 - s) : s;
    __syncthreads();                         // hbf from previous step visible

    // gh = h @ Whh^T via WMMA; B frags served straight from L0/L2
#pragma unroll
    for (int j = 0; j < 6; ++j) {
      const int tIdx = wave * 6 + j;         // 0..47
      const int mt = tIdx / 24, nt = tIdx % 24;
      v8f acc = {0.f,0.f,0.f,0.f,0.f,0.f,0.f,0.f};
#pragma unroll
      for (int kc = 0; kc < HD; kc += 32) {
        v16bf a = load_frag_rm(&hbf[mt * 16][0] + kc, HD, lane);
        v16bf b = load_frag_rm(Whh + nt * 16 * HD + kc, HD, lane);
        acc = __builtin_amdgcn_wmma_f32_16x16x32_bf16(false, a, false, b,
                                                      (short)0, acc, false, false);
      }
      const int n  = nt * 16 + (lane & 15);
      const int mB = mt * 16 + ((lane >> 4) << 3);
#pragma unroll
      for (int r = 0; r < 8; ++r) gh[mB + r][n] = acc[r];
    }
    __syncthreads();

    // gate update: 32x128 h-elements, 16 per thread (coalesced interleave)
#pragma unroll
    for (int i = 0; i < 16; ++i) {
      const int idx = tid + (i << 8);
      const int rowi = idx >> 7, c = idx & 127;
      const int seq = seq0 + rowi;
      if (seq < nSeq) {
        const long long tok = (long long)seq * seqStride +
                              (long long)step * stepStride;
        const float* xr = xi + tok * G3;
        if (i == 0 && s + 1 < seqLen) {
          const int nstep = reverse ? (seqLen - 2 - s) : (s + 1);
          __builtin_prefetch(xi + ((long long)seq * seqStride +
                                   (long long)nstep * stepStride) * G3 + c, 0, 1);
        }
        const float i_r = xr[c], i_z = xr[HD + c], i_n = xr[2 * HD + c];
        const float h_r = gh[rowi][c]          + bhh[c];
        const float h_z = gh[rowi][HD + c]     + bhh[HD + c];
        const float h_n = gh[rowi][2 * HD + c] + bhh[2 * HD + c];
        const float rg = 1.0f / (1.0f + __expf(-(i_r + h_r)));
        const float zg = 1.0f / (1.0f + __expf(-(i_z + h_z)));
        const float ng = tanhf(i_n + rg * h_n);
        const float hn = (1.0f - zg) * ng + zg * hreg[i];
        hreg[i] = hn;
        out[tok * outLd + outColOff + c] = hn;
        hbf[rowi][c] = (__bf16)hn;
      }
    }
  }
}

// ---------------------------------------------------------------------------
// Fused residual-add + LayerNorm over H=128, one wave32 per token.
// ---------------------------------------------------------------------------
__global__ __launch_bounds__(256)
void ln_kernel(const float* __restrict__ X, const float* __restrict__ R,
               const float* __restrict__ g, const float* __restrict__ b,
               float* __restrict__ Y, int ntok)
{
  const int gid = blockIdx.x * blockDim.x + threadIdx.x;
  const int t = gid >> 5, lane = gid & 31;
  if (t >= ntok) return;
  float4 xv = *(const float4*)(X + (long long)t * HD + lane * 4);
  if (R) {
    const float4 rv = *(const float4*)(R + (long long)t * HD + lane * 4);
    xv.x += rv.x; xv.y += rv.y; xv.z += rv.z; xv.w += rv.w;
  }
  float s = xv.x + xv.y + xv.z + xv.w;
#pragma unroll
  for (int m = 16; m > 0; m >>= 1) s += __shfl_xor(s, m, 32);
  const float mean = s * (1.0f / HD);
  const float d0 = xv.x - mean, d1 = xv.y - mean, d2 = xv.z - mean, d3 = xv.w - mean;
  float q = d0 * d0 + d1 * d1 + d2 * d2 + d3 * d3;
#pragma unroll
  for (int m = 16; m > 0; m >>= 1) q += __shfl_xor(q, m, 32);
  const float inv = rsqrtf(q * (1.0f / HD) + 1e-5f);
  const float4 gv = *(const float4*)(g + lane * 4);
  const float4 bv = *(const float4*)(b + lane * 4);
  float4 yv;
  yv.x = d0 * inv * gv.x + bv.x;
  yv.y = d1 * inv * gv.y + bv.y;
  yv.z = d2 * inv * gv.z + bv.z;
  yv.w = d3 * inv * gv.w + bv.w;
  *(float4*)(Y + (long long)t * HD + lane * 4) = yv;
}

// time_proj: out[p,c] = bt + sum_t Wt[t] * h[t,p,c]
__global__ void tproj_kernel(const float* __restrict__ h,
                             const float* __restrict__ Wt,
                             const float* __restrict__ bt,
                             float* __restrict__ o)
{
  const int idx = blockIdx.x * blockDim.x + threadIdx.x;
  if (idx >= PP * HD) return;
  float s = bt[0];
#pragma unroll
  for (int t = 0; t < TT; ++t) s += Wt[t] * h[(long long)t * PP * HD + idx];
  o[idx] = s;
}

__global__ void cvt_kernel(const float* __restrict__ s, __bf16* __restrict__ d, int n)
{
  const int i = blockIdx.x * blockDim.x + threadIdx.x;
  if (i < n) d[i] = (__bf16)s[i];
}

// ---------------------------------------------------------------------------
// Host orchestration
// ---------------------------------------------------------------------------
extern "C" void kernel_launch(void* const* d_in, const int* in_sizes, int n_in,
                              void* d_out, int out_size, void* d_ws, size_t ws_size,
                              hipStream_t stream)
{
  (void)in_sizes; (void)n_in; (void)out_size; (void)ws_size;
  const float* x = (const float*)d_in[0];
  auto F = [&](int i) { return (const float*)d_in[i]; };

  // workspace layout (fp32 unless noted)
  float* h    = (float*)d_ws;                      // [TOK][128]
  float* u    = h   + (long long)TOK * HD;         // [TOK][256] (also aliases pout)
  float* cat  = u   + (long long)TOK * DIN;        // [TOK][256] (also temporal out)
  float* xi   = cat + (long long)TOK * DIN;        // [TOK][384]
  float* tp   = xi  + (long long)TOK * G3;         // [PP][128]
  __bf16* wpool = (__bf16*)(tp + PP * HD);         // bf16 weight pool
  float* pout = u;
  float* tout = cat;

  long long woff = 0;
  auto cvtw = [&](int idx, int n) -> const __bf16* {
    __bf16* dst = wpool + woff; woff += n;
    cvt_kernel<<<(n + 255) / 256, 256, 0, stream>>>((const float*)d_in[idx], dst, n);
    return dst;
  };

  // d_in layout: 0:x 1:We 2:be 3:ge 4:bne then 2 blocks of 24 params, then
  // 53:Wt 54:bt 55:Wu 56:bu  (block: fwd{Wl,bl,Wih,Whh,bih,bhh} bwd{...}
  //                            Wp bp g1 b1 tmp{...} g2 b2)
  const __bf16* We = cvtw(1, HD * 320);
  struct MB { const __bf16 *Wl, *Wih, *Whh; const float *bl, *bih, *bhh; };
  struct Layer { MB fwd, bwd, tmp; const __bf16* Wp;
                 const float *bp, *g1, *b1, *g2, *b2; };
  Layer L[2];
  for (int l = 0; l < 2; ++l) {
    const int base = 5 + l * 24;
    auto mk = [&](int mb) {
      MB m;
      m.Wl  = cvtw(mb + 0, DIN * HD);
      m.bl  = F(mb + 1);
      m.Wih = cvtw(mb + 2, G3 * DIN);
      m.Whh = cvtw(mb + 3, G3 * HD);
      m.bih = F(mb + 4);
      m.bhh = F(mb + 5);
      return m;
    };
    L[l].fwd = mk(base + 0);
    L[l].bwd = mk(base + 6);
    L[l].Wp  = cvtw(base + 12, HD * DIN);
    L[l].bp  = F(base + 13);
    L[l].g1  = F(base + 14);
    L[l].b1  = F(base + 15);
    L[l].tmp = mk(base + 16);
    L[l].g2  = F(base + 22);
    L[l].b2  = F(base + 23);
  }
  const float*  Wt = F(53);
  const float*  bt = F(54);
  const __bf16* Wu = cvtw(55, 192 * HD);
  const float*  bu = F(56);

  const int M = TOK;
  dim3 blk(256);
  const int mg = (M + 63) / 64;
  const int lnGrid = (M * 32 + 255) / 256;

  // PatchEmbed: h = LN(x @ We^T + be)
  gemm_wmma_kernel<0><<<dim3(mg, 2), blk, 0, stream>>>(x, 320, We, 320, F(2), h, HD, M);
  ln_kernel<<<lnGrid, blk, 0, stream>>>(h, nullptr, F(3), F(4), h, M);

  for (int l = 0; l < 2; ++l) {
    // ---- spatial bidirectional (scan along P, batch = T=20 sequences) ----
    gemm_wmma_kernel<1><<<dim3(mg, 4), blk, 0, stream>>>(h, HD, L[l].fwd.Wl, HD, L[l].fwd.bl, u, DIN, M);
    gemm_wmma_kernel<0><<<dim3(mg, 6), blk, 0, stream>>>(u, DIN, L[l].fwd.Wih, DIN, L[l].fwd.bih, xi, G3, M);
    gru_scan_wmma<<<1, blk, 0, stream>>>(xi, L[l].fwd.Whh, L[l].fwd.bhh,
                                         cat, DIN, 0, TT, PP, 1, PP, 0);
    gemm_wmma_kernel<1><<<dim3(mg, 4), blk, 0, stream>>>(h, HD, L[l].bwd.Wl, HD, L[l].bwd.bl, u, DIN, M);
    gemm_wmma_kernel<0><<<dim3(mg, 6), blk, 0, stream>>>(u, DIN, L[l].bwd.Wih, DIN, L[l].bwd.bih, xi, G3, M);
    gru_scan_wmma<<<1, blk, 0, stream>>>(xi, L[l].bwd.Whh, L[l].bwd.bhh,
                                         cat, DIN, HD, TT, PP, 1, PP, 1);
    gemm_wmma_kernel<0><<<dim3(mg, 2), blk, 0, stream>>>(cat, DIN, L[l].Wp, DIN, L[l].bp, pout, HD, M);
    ln_kernel<<<lnGrid, blk, 0, stream>>>(pout, h, L[l].g1, L[l].b1, h, M);

    // ---- temporal causal (scan along T, batch = P=5000 sequences) ----
    gemm_wmma_kernel<1><<<dim3(mg, 4), blk, 0, stream>>>(h, HD, L[l].tmp.Wl, HD, L[l].tmp.bl, u, DIN, M);
    gemm_wmma_kernel<0><<<dim3(mg, 6), blk, 0, stream>>>(u, DIN, L[l].tmp.Wih, DIN, L[l].tmp.bih, xi, G3, M);
    gru_scan_wmma<<<(PP + 31) / 32, blk, 0, stream>>>(xi, L[l].tmp.Whh, L[l].tmp.bhh,
                                                      tout, HD, 0, PP, TT, PP, 1, 0);
    ln_kernel<<<lnGrid, blk, 0, stream>>>(tout, h, L[l].g2, L[l].b2, h, M);
  }

  // time_proj (T->1) then PatchUnembed (H -> S*3 = 192)
  tproj_kernel<<<(PP * HD + 255) / 256, blk, 0, stream>>>(h, Wt, bt, tp);
  gemm_wmma_kernel<0><<<dim3((PP + 63) / 64, 3), blk, 0, stream>>>(tp, HD, Wu, HD, bu,
                                                                   (float*)d_out, 192, PP);
}
